// KVAE_71373766525697
// MI455X (gfx1250) — matrix operand order repaired and checked by reading, not compiled, not run on hardware
//
#include <hip/hip_runtime.h>

typedef _Float16 half_t;
typedef __attribute__((ext_vector_type(16))) _Float16 v16h;
typedef __attribute__((ext_vector_type(8)))  float    v8f;
typedef __attribute__((ext_vector_type(4)))  unsigned int uint4v;

#define T_LEN  256
#define BATCH  256
#define M_ROWS 65536
#define X_DIM  513
#define XP     544
#define NPGEN  528   /* gen_W output dim 513 padded to 16-row multiple */
#define DH     256
#define A_DIM  16
#define H_RNN  128
#define GATES  512
#define Z_DIM  8

union Frag16 {
  v16h   m;
  uint4v q[2];
  half_t e[16];
};

__device__ inline v8f wmma_f16(const Frag16& a, const Frag16& b, v8f c) {
  // D = A(16x32 f16) * B(32x16 f16) + C(16x16 f32)
  return __builtin_amdgcn_wmma_f32_16x16x32_f16(false, a.m, false, b.m, (short)0, c,
                                                false, false);
}

__device__ inline float sigmoidf_(float x) { return 1.f / (1.f + expf(-x)); }

// ---------------------------------------------------------------------------
// x (B, X_DIM, T) f32  ->  XT (T*B, XP) f16 row-major, zero-padded K
// ---------------------------------------------------------------------------
__global__ __launch_bounds__(256) void k_transpose_x(const float* __restrict__ x,
                                                     half_t* __restrict__ xt) {
  size_t idx = (size_t)blockIdx.x * 256 + threadIdx.x;
  if (idx >= (size_t)M_ROWS * XP) return;
  int k = (int)(idx % XP);
  int r = (int)(idx / XP);
  int t = r / BATCH, b = r % BATCH;
  float v = 0.f;
  if (k < X_DIM) v = x[((size_t)b * X_DIM + k) * T_LEN + t];
  xt[idx] = (half_t)v;
}

// ---------------------------------------------------------------------------
// Weight prep: src f32 (KxN, or NxK if nk=1) -> dst f16 (Npad x Kpad),
// zero-padded in both K and N so GEMM fragment loads never need guards.
// ---------------------------------------------------------------------------
__global__ __launch_bounds__(256) void k_prep_w(const float* __restrict__ src,
                                                half_t* __restrict__ dst,
                                                int K, int N, int Kpad, int Npad,
                                                int nk) {
  int idx = blockIdx.x * 256 + threadIdx.x;
  if (idx >= Npad * Kpad) return;
  int k = idx % Kpad, n = idx / Kpad;
  float v = 0.f;
  if (k < K && n < N) v = nk ? src[(size_t)n * K + k] : src[(size_t)k * N + n];
  dst[idx] = (half_t)v;
}

// ---------------------------------------------------------------------------
// WMMA GEMM: out = act(A(M x Kpad f16) @ BT^T + bias). BT is (Npad x ldb) f16,
// zero-padded, so the k-loop has no conditionals. Each wave computes TWO
// 16x16 m-tiles sharing one B fragment (halves B traffic, 2 independent
// WMMAs per k-step for latency hiding). act: 0=none 1=tanh 2=exp
// ---------------------------------------------------------------------------
__global__ __launch_bounds__(256) void k_gemm(const half_t* __restrict__ A, int lda,
                                              const half_t* __restrict__ BT, int ldb,
                                              int Neff,
                                              const float* __restrict__ bias,
                                              float* __restrict__ outF,
                                              half_t* __restrict__ outH, int ldo,
                                              int Kpad, int act) {
  int wave = threadIdx.x >> 5, lane = threadIdx.x & 31;
  int m0 = blockIdx.x * 32;
  int n0 = (blockIdx.y * 8 + wave) * 16;
  if (n0 >= Neff) return;                 // wave-uniform
  int lo = lane & 15, hi = lane >> 4;
  int colB = n0 + lo;                     // < Npad always (tile-aligned pad)
  const half_t* ap0 = A + (size_t)(m0 + lo) * lda;
  const half_t* ap1 = A + (size_t)(m0 + 16 + lo) * lda;
  const half_t* bp  = BT + (size_t)colB * ldb;
  v8f acc0, acc1;
  for (int i = 0; i < 8; ++i) { acc0[i] = 0.f; acc1[i] = 0.f; }
  for (int k0 = 0; k0 < Kpad; k0 += 32) {
    Frag16 a0, a1, bf;
    // B frag: lane lo=col, element e -> K = k0 + hi*16 + e
    bf.q[0] = *(const uint4v*)(bp + k0 + hi * 16);
    bf.q[1] = *(const uint4v*)(bp + k0 + hi * 16 + 8);
    // A frag: lane lo=row, element e -> K = k0 + hi*16 + e
    a0.q[0] = *(const uint4v*)(ap0 + k0 + hi * 16);
    a0.q[1] = *(const uint4v*)(ap0 + k0 + hi * 16 + 8);
    a1.q[0] = *(const uint4v*)(ap1 + k0 + hi * 16);
    a1.q[1] = *(const uint4v*)(ap1 + k0 + hi * 16 + 8);
    __builtin_prefetch(ap0 + k0 + 128, 0, 0);   // global_prefetch_b8 (speculative)
    __builtin_prefetch(ap1 + k0 + 128, 0, 0);
    acc0 = wmma_f16(a0, bf, acc0);
    acc1 = wmma_f16(a1, bf, acc1);
  }
  if (colB >= Neff) return;               // store guard only (lane-divergent ok)
  float bs = bias ? bias[colB] : 0.f;
  for (int v = 0; v < 8; ++v) {
    int ma = m0 + v + hi * 8;             // C/D layout: M = vgpr + 8*hi, N = lo
    int mb = ma + 16;
    float va = acc0[v] + bs;
    float vb = acc1[v] + bs;
    if (act == 1) { va = tanhf(va); vb = tanhf(vb); }
    else if (act == 2) { va = expf(va); vb = expf(vb); }
    if (outF) {
      outF[(size_t)ma * ldo + colB] = va;
      outF[(size_t)mb * ldo + colB] = vb;
    }
    if (outH) {
      outH[(size_t)ma * ldo + colB] = (half_t)va;
      outH[(size_t)mb * ldo + colB] = (half_t)vb;
    }
  }
}

// ---------------------------------------------------------------------------
// a = mean + eps*exp(0.5*logvar); also f16 copy (stride 32, padded) and
// time-shifted a_tm1 (row 0..B-1 = a_init)
// ---------------------------------------------------------------------------
__global__ __launch_bounds__(256) void k_make_a(const float* __restrict__ mean,
                                                const float* __restrict__ logv,
                                                const float* __restrict__ eps,
                                                const float* __restrict__ ainit,
                                                float* __restrict__ a32,
                                                half_t* __restrict__ a16,
                                                half_t* __restrict__ ash16) {
  int idx = blockIdx.x * 256 + threadIdx.x;
  if (idx >= M_ROWS * A_DIM) return;
  int r = idx >> 4, i = idx & 15;
  float av = mean[idx] + eps[idx] * expf(0.5f * logv[idx]);
  a32[idx] = av;
  a16[(size_t)r * 32 + i]      = (half_t)av;
  a16[(size_t)r * 32 + 16 + i] = (half_t)0.f;
  int rn = r + BATCH;
  if (rn < M_ROWS) {
    ash16[(size_t)rn * 32 + i]      = (half_t)av;
    ash16[(size_t)rn * 32 + 16 + i] = (half_t)0.f;
  }
  if (r < BATCH) {
    ash16[(size_t)r * 32 + i]      = (half_t)ainit[i];
    ash16[(size_t)r * 32 + 16 + i] = (half_t)0.f;
  }
}

// ---------------------------------------------------------------------------
// Persistent single-WGP LSTM. 32 waves; h state f16 in LDS (64KB); WMMA gates.
// Emits alpha = softmax(h @ alphaW + b) per step.
// ---------------------------------------------------------------------------
__global__ __launch_bounds__(1024) void k_lstm(const half_t* __restrict__ ash16,
                                               const half_t* __restrict__ WihT,
                                               const half_t* __restrict__ WhhT,
                                               const float* __restrict__ bih,
                                               const float* __restrict__ bhh,
                                               const float* __restrict__ alphaW,
                                               const float* __restrict__ alphaB,
                                               float* __restrict__ gates,
                                               float* __restrict__ cstate,
                                               float* __restrict__ alphaOut) {
  extern __shared__ char smem[];
  half_t* hlds = (half_t*)smem;           // BATCH*H_RNN f16 = 64KB
  int tid = threadIdx.x, wave = tid >> 5, lane = tid & 31;
  int lo = lane & 15, hi = lane >> 4;
  for (int i = tid; i < BATCH * H_RNN; i += 1024) {
    hlds[i] = (half_t)0.f;
    cstate[i] = 0.f;
  }
  __syncthreads();
  for (int t = 0; t < T_LEN; ++t) {
    // ---- phase 1: gates = a_tm1 @ Wih^T + h @ Whh^T + b (WMMA) ----
    for (int it = 0; it < 16; ++it) {
      int tile = wave * 16 + it;          // 512 tiles = 16 mt x 32 nt
      int m0 = (tile >> 5) * 16, n0 = (tile & 31) * 16;
      int row = m0 + lo, colB = n0 + lo;
      v8f acc; for (int i = 0; i < 8; ++i) acc[i] = 0.f;
      Frag16 af, bf;
      // a part (K=16, padded to 32 with zeros in storage)
      const half_t* ap = ash16 + ((size_t)t * BATCH + row) * 32;
      af.q[0] = *(const uint4v*)(ap + hi * 16);
      af.q[1] = *(const uint4v*)(ap + hi * 16 + 8);
      const half_t* bp = WihT + (size_t)colB * 32 + hi * 16;
      bf.q[0] = *(const uint4v*)(bp);
      bf.q[1] = *(const uint4v*)(bp + 8);
      acc = wmma_f16(af, bf, acc);
      // h part (K=128), A from LDS
      for (int kc = 0; kc < 4; ++kc) {
        int k0 = kc * 32;
        const half_t* hp = hlds + row * H_RNN + k0;
        af.q[0] = *(const uint4v*)(hp + hi * 16);
        af.q[1] = *(const uint4v*)(hp + hi * 16 + 8);
        const half_t* wp = WhhT + (size_t)colB * H_RNN + k0 + hi * 16;
        bf.q[0] = *(const uint4v*)(wp);
        bf.q[1] = *(const uint4v*)(wp + 8);
        acc = wmma_f16(af, bf, acc);
      }
      float bs = bih[colB] + bhh[colB];
      for (int v = 0; v < 8; ++v) {
        int m = m0 + v + hi * 8;
        gates[(size_t)m * GATES + colB] = acc[v] + bs;
      }
    }
    __threadfence();
    __syncthreads();
    // ---- phase 2: c,h update ----
    for (int idx = tid; idx < BATCH * H_RNN; idx += 1024) {
      int b = idx >> 7, j = idx & 127;
      const float* g = gates + (size_t)b * GATES;
      float c = cstate[idx];
      c = sigmoidf_(g[128 + j]) * c + sigmoidf_(g[j]) * tanhf(g[256 + j]);
      float h = sigmoidf_(g[384 + j]) * tanhf(c);
      cstate[idx] = c;
      hlds[idx] = (half_t)h;
    }
    __syncthreads();
    // ---- phase 3: alpha = softmax(h @ alphaW + b) ----
    if (tid < BATCH) {
      float s0 = alphaB[0], s1 = alphaB[1], s2 = alphaB[2];
      const half_t* hb = hlds + tid * H_RNN;
      for (int j = 0; j < H_RNN; ++j) {
        float h = (float)hb[j];
        s0 += h * alphaW[j * 3 + 0];
        s1 += h * alphaW[j * 3 + 1];
        s2 += h * alphaW[j * 3 + 2];
      }
      float mx = fmaxf(s0, fmaxf(s1, s2));
      float e0 = expf(s0 - mx), e1 = expf(s1 - mx), e2 = expf(s2 - mx);
      float inv = 1.f / (e0 + e1 + e2);
      float* ao = alphaOut + ((size_t)t * BATCH + tid) * 3;
      ao[0] = e0 * inv; ao[1] = e1 * inv; ao[2] = e2 * inv;
    }
    __syncthreads();
  }
}

// ---------------------------------------------------------------------------
// Kalman filter + RTS smoother: one block per batch element, 64 threads.
// ---------------------------------------------------------------------------
__global__ __launch_bounds__(64) void k_kalman(const float* __restrict__ alpha,
                                               const float* __restrict__ a32,
                                               const float* __restrict__ ainit,
                                               const float* __restrict__ Ap,
                                               const float* __restrict__ Bp,
                                               const float* __restrict__ Cp,
                                               float* __restrict__ muF,
                                               float* __restrict__ SigF,
                                               float* __restrict__ SigP,
                                               float* __restrict__ agen) {
  __shared__ float sAl[6];
  __shared__ float sCt[128], sAt[64], sBt[128];
  __shared__ float sMu[8], sMuP[8], sSig[64], sSigP[64];
  __shared__ float sRes[16], sAT[16], sU[16];
  __shared__ float sCS[128], sS[256], sAug[512], sCol[16];
  __shared__ float sKt[128], sIKC[64], sT1[64], sT2[64];
  __shared__ float sMuS[8], sMuN[8], sMuFt[8], sMuNw[8];
  __shared__ float sSigS[64], sSigFt[64], sPn[64], sD[64], sJ[64], sSigNw[64];
  int b = blockIdx.x, tid = threadIdx.x;

  // ---------------- forward filter ----------------
  for (int t = 0; t < T_LEN; ++t) {
    if (tid < 3) sAl[tid] = alpha[((size_t)t * BATCH + b) * 3 + tid];
    __syncthreads();
    for (int i = tid; i < 128; i += 64) {
      sCt[i] = sAl[0] * Cp[i] + sAl[1] * Cp[128 + i] + sAl[2] * Cp[256 + i];
      sBt[i] = sAl[0] * Bp[i] + sAl[1] * Bp[128 + i] + sAl[2] * Bp[256 + i];
    }
    sAt[tid] = sAl[0] * Ap[tid] + sAl[1] * Ap[64 + tid] + sAl[2] * Ap[128 + tid];
    if (tid < 16) {
      sAT[tid] = a32[((size_t)t * BATCH + b) * 16 + tid];
      sU[tid]  = (t > 0) ? a32[((size_t)(t - 1) * BATCH + b) * 16 + tid] : ainit[tid];
    }
    __syncthreads();
    // prediction
    if (t == 0) {
      if (tid < 8) sMuP[tid] = 0.f;
      sSigP[tid] = ((tid % 9) == 0) ? 20.f : 0.f;
    } else {
      if (tid < 8) {
        float s = 0.f;
        for (int j = 0; j < 8; ++j)  s += sAt[tid * 8 + j] * sMu[j];
        for (int j = 0; j < 16; ++j) s += sBt[tid * 16 + j] * sU[j];
        sMuP[tid] = s;
      }
      {
        int r = tid >> 3, c = tid & 7;
        float s = 0.f;
        for (int k = 0; k < 8; ++k) s += sAt[r * 8 + k] * sSig[k * 8 + c];
        sT1[tid] = s;
      }
      __syncthreads();
      {
        int r = tid >> 3, c = tid & 7;
        float s = (r == c) ? 0.08f : 0.f;             // Q
        for (int k = 0; k < 8; ++k) s += sT1[r * 8 + k] * sAt[c * 8 + k];
        sSigP[tid] = s;
      }
    }
    __syncthreads();
    // measurement: res, CS = C*SigP, S = CS*C^T + R
    if (tid < 16) {
      float s = 0.f;
      for (int j = 0; j < 8; ++j) s += sCt[tid * 8 + j] * sMuP[j];
      sRes[tid] = sAT[tid] - s;
    }
    for (int i = tid; i < 128; i += 64) {
      int r = i >> 3, c = i & 7;
      float s = 0.f;
      for (int k = 0; k < 8; ++k) s += sCt[r * 8 + k] * sSigP[k * 8 + c];
      sCS[i] = s;
    }
    __syncthreads();
    for (int i = tid; i < 256; i += 64) {
      int r = i >> 4, c = i & 15;
      float s = (r == c) ? 0.03f : 0.f;               // R
      for (int k = 0; k < 8; ++k) s += sCS[r * 8 + k] * sCt[c * 8 + k];
      sS[i] = s;
    }
    __syncthreads();
    // invert S (16x16 Gauss-Jordan)
    for (int i = tid; i < 512; i += 64) {
      int r = i >> 5, c = i & 31;
      sAug[i] = (c < 16) ? sS[r * 16 + c] : ((c - 16 == r) ? 1.f : 0.f);
    }
    __syncthreads();
    for (int p = 0; p < 16; ++p) {
      if (tid < 16) sCol[tid] = sAug[tid * 32 + p];
      __syncthreads();
      float pinv = 1.f / sCol[p];
      if (tid < 32) sAug[p * 32 + tid] *= pinv;
      __syncthreads();
      for (int i = tid; i < 512; i += 64) {
        int r = i >> 5, c = i & 31;
        if (r != p) sAug[i] -= sCol[r] * sAug[p * 32 + c];
      }
      __syncthreads();
    }
    // Kt = (C SigP)^T Sinv  (SigP symmetric)
    for (int i = tid; i < 128; i += 64) {
      int r = i >> 4, c = i & 15;
      float s = 0.f;
      for (int k = 0; k < 16; ++k) s += sCS[k * 8 + r] * sAug[k * 32 + 16 + c];
      sKt[i] = s;
    }
    __syncthreads();
    if (tid < 8) {
      float s = 0.f;
      for (int j = 0; j < 16; ++j) s += sKt[tid * 16 + j] * sRes[j];
      sMu[tid] = sMuP[tid] + s;
    }
    {
      int r = tid >> 3, c = tid & 7;
      float s = (r == c) ? 1.f : 0.f;
      for (int k = 0; k < 16; ++k) s -= sKt[r * 16 + k] * sCt[k * 8 + c];
      sIKC[tid] = s;
    }
    __syncthreads();
    {
      int r = tid >> 3, c = tid & 7;
      float s = 0.f;
      for (int k = 0; k < 8; ++k) s += sIKC[r * 8 + k] * sSigP[k * 8 + c];
      sT1[tid] = s;
    }
    __syncthreads();
    {
      int r = tid >> 3, c = tid & 7;
      float s = 0.f, s2 = 0.f;
      for (int k = 0; k < 8; ++k)  s  += sT1[r * 8 + k] * sIKC[c * 8 + k];
      for (int k = 0; k < 16; ++k) s2 += sKt[r * 16 + k] * sKt[c * 16 + k];
      sSig[tid] = s + 0.03f * s2;                     // + Kt R Kt^T
    }
    __syncthreads();
    if (tid < 8) muF[((size_t)t * BATCH + b) * 8 + tid] = sMu[tid];
    SigF[((size_t)t * BATCH + b) * 64 + tid] = sSig[tid];
    SigP[((size_t)t * BATCH + b) * 64 + tid] = sSigP[tid];
    __syncthreads();
  }

  // ---------------- backward smoother ----------------
  if (tid < 8) sMuS[tid] = sMu[tid];
  sSigS[tid] = sSig[tid];
  if (tid < 16) {                                     // a_gen[T-1] = C_{T-1} mu_f[T-1]
    float s = 0.f;
    for (int j = 0; j < 8; ++j) s += sCt[tid * 8 + j] * sMu[j];
    agen[((size_t)(T_LEN - 1) * BATCH + b) * 16 + tid] = s;
  }
  __syncthreads();
  for (int t = T_LEN - 2; t >= 0; --t) {
    if (tid < 3) {
      sAl[tid]     = alpha[((size_t)(t + 1) * BATCH + b) * 3 + tid];
      sAl[3 + tid] = alpha[((size_t)t * BATCH + b) * 3 + tid];
    }
    if (tid < 8) {
      sMuFt[tid] = muF[((size_t)t * BATCH + b) * 8 + tid];
      sMuN[tid]  = muF[((size_t)(t + 1) * BATCH + b) * 8 + tid];
    }
    sSigFt[tid] = SigF[((size_t)t * BATCH + b) * 64 + tid];
    sPn[tid]    = SigP[((size_t)(t + 1) * BATCH + b) * 64 + tid];
    __syncthreads();
    sAt[tid] = sAl[0] * Ap[tid] + sAl[1] * Ap[64 + tid] + sAl[2] * Ap[128 + tid];
    for (int i = tid; i < 128; i += 64)
      sCt[i] = sAl[3] * Cp[i] + sAl[4] * Cp[128 + i] + sAl[5] * Cp[256 + i];
    for (int i = tid; i < 128; i += 64) {
      int r = i >> 4, c = i & 15;
      sAug[i] = (c < 8) ? sPn[r * 8 + c] : ((c - 8 == r) ? 1.f : 0.f);
    }
    __syncthreads();
    for (int p = 0; p < 8; ++p) {                     // invert SigP_{t+1} (8x8)
      if (tid < 8) sCol[tid] = sAug[tid * 16 + p];
      __syncthreads();
      float pinv = 1.f / sCol[p];
      if (tid < 16) sAug[p * 16 + tid] *= pinv;
      __syncthreads();
      for (int i = tid; i < 128; i += 64) {
        int r = i >> 4, c = i & 15;
        if (r != p) sAug[i] -= sCol[r] * sAug[p * 16 + c];
      }
      __syncthreads();
    }
    {
      int r = tid >> 3, c = tid & 7;                  // SigF @ A^T
      float s = 0.f;
      for (int k = 0; k < 8; ++k) s += sSigFt[r * 8 + k] * sAt[c * 8 + k];
      sT1[tid] = s;
    }
    __syncthreads();
    {
      int r = tid >> 3, c = tid & 7;                  // J = (SigF A^T) Pinv
      float s = 0.f;
      for (int k = 0; k < 8; ++k) s += sT1[r * 8 + k] * sAug[k * 16 + 8 + c];
      sJ[tid] = s;
      sD[tid] = sSigS[tid] - sPn[tid];
    }
    __syncthreads();
    if (tid < 8) {
      float s = 0.f;
      for (int j = 0; j < 8; ++j) s += sJ[tid * 8 + j] * (sMuS[j] - sMuN[j]);
      sMuNw[tid] = sMuFt[tid] + s;
    }
    {
      int r = tid >> 3, c = tid & 7;
      float s = 0.f;
      for (int k = 0; k < 8; ++k) s += sJ[r * 8 + k] * sD[k * 8 + c];
      sT2[tid] = s;
    }
    __syncthreads();
    {
      int r = tid >> 3, c = tid & 7;
      float s = 0.f;
      for (int k = 0; k < 8; ++k) s += sT2[r * 8 + k] * sJ[c * 8 + k];
      sSigNw[tid] = sSigFt[tid] + s;
    }
    __syncthreads();
    if (tid < 8) sMuS[tid] = sMuNw[tid];
    sSigS[tid] = sSigNw[tid];
    __syncthreads();
    if (tid < 16) {
      float s = 0.f;
      for (int j = 0; j < 8; ++j) s += sCt[tid * 8 + j] * sMuS[j];
      agen[((size_t)t * BATCH + b) * 16 + tid] = s;
    }
    __syncthreads();
  }
}

// ---------------------------------------------------------------------------
// Workspace layout (bytes). Kalman buffers alias the XT region (free by then).
// ---------------------------------------------------------------------------
#define OFF_XT     0ULL
#define OFF_H0     (OFF_XT    + (size_t)M_ROWS * XP * 2)
#define OFF_H1     (OFF_H0    + (size_t)M_ROWS * DH * 2)
#define OFF_MEAN   (OFF_H1    + (size_t)M_ROWS * DH * 2)
#define OFF_LOGV   (OFF_MEAN  + (size_t)M_ROWS * A_DIM * 4)
#define OFF_A32    (OFF_LOGV  + (size_t)M_ROWS * A_DIM * 4)
#define OFF_A16    (OFF_A32   + (size_t)M_ROWS * A_DIM * 4)
#define OFF_ASH    (OFF_A16   + (size_t)M_ROWS * 32 * 2)
#define OFF_WENC0  (OFF_ASH   + (size_t)M_ROWS * 32 * 2)
#define OFF_WENC1  (OFF_WENC0 + (size_t)DH * XP * 2)
#define OFF_WIM    (OFF_WENC1 + (size_t)DH * DH * 2)
#define OFF_WIL    (OFF_WIM   + (size_t)A_DIM * DH * 2)
#define OFF_WDEC0  (OFF_WIL   + (size_t)A_DIM * DH * 2)
#define OFF_WDEC1  (OFF_WDEC0 + (size_t)DH * 32 * 2)
#define OFF_WGEN   (OFF_WDEC1 + (size_t)DH * DH * 2)
#define OFF_WIH    (OFF_WGEN  + (size_t)NPGEN * DH * 2)
#define OFF_WHH    (OFF_WIH   + (size_t)GATES * 32 * 2)
#define OFF_GATES  (OFF_WHH   + (size_t)GATES * H_RNN * 2)
#define OFF_CST    (OFF_GATES + (size_t)BATCH * GATES * 4)
#define OFF_ALPHA  (OFF_CST   + (size_t)BATCH * H_RNN * 4)
#define OFF_MUF    0ULL
#define OFF_SIGF   (OFF_MUF  + (size_t)M_ROWS * Z_DIM * 4)
#define OFF_SIGP   (OFF_SIGF + (size_t)M_ROWS * Z_DIM * Z_DIM * 4)

extern "C" void kernel_launch(void* const* d_in, const int* in_sizes, int n_in,
                              void* d_out, int out_size, void* d_ws, size_t ws_size,
                              hipStream_t stream) {
  (void)in_sizes; (void)n_in; (void)out_size; (void)ws_size;
  const float* x     = (const float*)d_in[0];
  const float* eps   = (const float*)d_in[1];
  const float* encW0 = (const float*)d_in[2];
  const float* encB0 = (const float*)d_in[3];
  const float* encW1 = (const float*)d_in[4];
  const float* encB1 = (const float*)d_in[5];
  const float* imW   = (const float*)d_in[6];
  const float* imB   = (const float*)d_in[7];
  const float* ilW   = (const float*)d_in[8];
  const float* ilB   = (const float*)d_in[9];
  const float* decW0 = (const float*)d_in[10];
  const float* decB0 = (const float*)d_in[11];
  const float* decW1 = (const float*)d_in[12];
  const float* decB1 = (const float*)d_in[13];
  const float* genW  = (const float*)d_in[14];
  const float* genB  = (const float*)d_in[15];
  const float* Wih   = (const float*)d_in[16];
  const float* Whh   = (const float*)d_in[17];
  const float* bih   = (const float*)d_in[18];
  const float* bhh   = (const float*)d_in[19];
  const float* alW   = (const float*)d_in[20];
  const float* alB   = (const float*)d_in[21];
  const float* Ap    = (const float*)d_in[22];
  const float* Bp    = (const float*)d_in[23];
  const float* Cp    = (const float*)d_in[24];
  const float* ainit = (const float*)d_in[25];

  char* ws = (char*)d_ws;
  half_t* XT    = (half_t*)(ws + OFF_XT);
  half_t* H0    = (half_t*)(ws + OFF_H0);     // reused as G0
  half_t* H1    = (half_t*)(ws + OFF_H1);     // reused as G1
  float*  MEAN  = (float*)(ws + OFF_MEAN);
  float*  LOGV  = (float*)(ws + OFF_LOGV);
  float*  A32   = (float*)(ws + OFF_A32);
  half_t* A16h  = (half_t*)(ws + OFF_A16);
  half_t* ASH   = (half_t*)(ws + OFF_ASH);
  half_t* WET0  = (half_t*)(ws + OFF_WENC0);
  half_t* WET1  = (half_t*)(ws + OFF_WENC1);
  half_t* WIMT  = (half_t*)(ws + OFF_WIM);
  half_t* WILT  = (half_t*)(ws + OFF_WIL);
  half_t* WDT0  = (half_t*)(ws + OFF_WDEC0);
  half_t* WDT1  = (half_t*)(ws + OFF_WDEC1);
  half_t* WGT   = (half_t*)(ws + OFF_WGEN);
  half_t* WIHT  = (half_t*)(ws + OFF_WIH);
  half_t* WHHT  = (half_t*)(ws + OFF_WHH);
  float*  GWS   = (float*)(ws + OFF_GATES);
  float*  CST   = (float*)(ws + OFF_CST);
  float*  ALPHA = (float*)(ws + OFF_ALPHA);
  float*  MUF   = (float*)(ws + OFF_MUF);
  float*  SIGFW = (float*)(ws + OFF_SIGF);
  float*  SIGPW = (float*)(ws + OFF_SIGP);
  float*  Y_OUT = (float*)d_out;
  float*  AGEN  = (float*)d_out + (size_t)M_ROWS * X_DIM;

  auto nb = [](size_t n) { return dim3((unsigned)((n + 255) / 256)); };

  // weight prep (f16, N-major, zero-padded in K and N)
  k_prep_w<<<nb((size_t)DH * XP), 256, 0, stream>>>(encW0, WET0, X_DIM, DH, XP, DH, 0);
  k_prep_w<<<nb((size_t)DH * DH), 256, 0, stream>>>(encW1, WET1, DH, DH, DH, DH, 0);
  k_prep_w<<<nb((size_t)A_DIM * DH), 256, 0, stream>>>(imW, WIMT, DH, A_DIM, DH, A_DIM, 0);
  k_prep_w<<<nb((size_t)A_DIM * DH), 256, 0, stream>>>(ilW, WILT, DH, A_DIM, DH, A_DIM, 0);
  k_prep_w<<<nb((size_t)DH * 32), 256, 0, stream>>>(decW0, WDT0, A_DIM, DH, 32, DH, 0);
  k_prep_w<<<nb((size_t)DH * DH), 256, 0, stream>>>(decW1, WDT1, DH, DH, DH, DH, 0);
  k_prep_w<<<nb((size_t)NPGEN * DH), 256, 0, stream>>>(genW, WGT, DH, X_DIM, DH, NPGEN, 0);
  k_prep_w<<<nb((size_t)GATES * 32), 256, 0, stream>>>(Wih, WIHT, A_DIM, GATES, 32, GATES, 1);
  k_prep_w<<<nb((size_t)GATES * H_RNN), 256, 0, stream>>>(Whh, WHHT, H_RNN, GATES, H_RNN, GATES, 1);
  k_transpose_x<<<nb((size_t)M_ROWS * XP), 256, 0, stream>>>(x, XT);

  // encoder / decoder GEMMs (each wave: 2 m-tiles of 16x16)
  dim3 g256(M_ROWS / 32, 2);   // 16 n-tiles / 8 waves
  dim3 g16(M_ROWS / 32, 1);    // 1 n-tile
  dim3 g513(M_ROWS / 32, 5);   // 33 n-tiles
  k_gemm<<<g256, 256, 0, stream>>>(XT, XP, WET0, XP, DH, encB0, nullptr, H0, DH, XP, 1);
  k_gemm<<<g256, 256, 0, stream>>>(H0, DH, WET1, DH, DH, encB1, nullptr, H1, DH, DH, 1);
  k_gemm<<<g16, 256, 0, stream>>>(H1, DH, WIMT, DH, A_DIM, imB, MEAN, nullptr, A_DIM, DH, 0);
  k_gemm<<<g16, 256, 0, stream>>>(H1, DH, WILT, DH, A_DIM, ilB, LOGV, nullptr, A_DIM, DH, 0);
  k_make_a<<<nb((size_t)M_ROWS * A_DIM), 256, 0, stream>>>(MEAN, LOGV, eps, ainit,
                                                           A32, A16h, ASH);
  k_gemm<<<g256, 256, 0, stream>>>(A16h, 32, WDT0, 32, DH, decB0, nullptr, H0, DH, 32, 1);
  k_gemm<<<g256, 256, 0, stream>>>(H0, DH, WDT1, DH, DH, decB1, nullptr, H1, DH, DH, 1);
  k_gemm<<<g513, 256, 0, stream>>>(H1, DH, WGT, DH, X_DIM, genB, Y_OUT, nullptr, X_DIM, DH, 2);

  // LSTM (single WGP, h-state in LDS) + alpha
  k_lstm<<<1, 1024, BATCH * H_RNN * sizeof(half_t), stream>>>(
      ASH, WIHT, WHHT, bih, bhh, alW, alB, GWS, CST, ALPHA);

  // Kalman filter + RTS smoother + a_gen
  k_kalman<<<BATCH, 64, 0, stream>>>(ALPHA, A32, ainit, Ap, Bp, Cp,
                                     MUF, SIGFW, SIGPW, AGEN);
}
